// PeriodicKernelAttention_82592221102308
// MI455X (gfx1250) — matrix operand order repaired and checked by reading, not compiled, not run on hardware
//
#include <hip/hip_runtime.h>
#include <hip/hip_bf16.h>

typedef __attribute__((ext_vector_type(16))) _Float16 v16h;
typedef __attribute__((ext_vector_type(8)))  _Float16 v8h;   // 16-byte fragment half
typedef __attribute__((ext_vector_type(8)))  float    v8f;

#define S_LEN 2048
#define HEAD_D 64
#define BH_CNT 32              // 4 batches * 8 heads
#define NROWS (BH_CNT * S_LEN) // 65536 rows of Q (and of K, V)
#define QTILE 32               // q rows per workgroup
#define LDS_STRIDE 2052        // 2048 + 4: stride % 64 == 4 -> conflict-free col reads
#define TWO_APPROX 2.00001f

// Assemble a 16x32-f16 A fragment (ISA 7.12.2): elements [0..7] = 16 contiguous
// bytes at halfs [base+8*half, +8), elements [8..15] at [base+16+8*half, +8).
__device__ __forceinline__ v16h load_fragA(const _Float16* base, int half) {
    const v8h lo = *(const v8h*)(base + 8 * half);        // one b128
    const v8h hi = *(const v8h*)(base + 16 + 8 * half);   // one b128
    v16h a;
    #pragma unroll
    for (int v = 0; v < 8; ++v) { a[v] = lo[v]; a[8 + v] = hi[v]; }
    return a;
}
// 32x16-f16 B fragment: elements [0..7] at halfs [base+8*half, +8),
// elements [8..15] at [base+16+8*half, +8)  (same K order as load_fragA).
__device__ __forceinline__ v16h load_fragB(const _Float16* base, int half) {
    const v8h lo = *(const v8h*)(base + 8 * half);        // b128
    const v8h hi = *(const v8h*)(base + 16 + 8 * half);   // b128
    v16h b;
    #pragma unroll
    for (int v = 0; v < 8; ++v) { b[v] = lo[v]; b[8 + v] = hi[v]; }
    return b;
}

// ---------------------------------------------------------------------------
// Kernel 0: per-row L2 normalization of Q or K -> f16, plus sum-of-squares.
// ---------------------------------------------------------------------------
__global__ __launch_bounds__(256)
void pka_norm_precompute(const float* __restrict__ src,
                         _Float16* __restrict__ dst,
                         float* __restrict__ sq_out) {
    const int wave = threadIdx.x >> 5;
    const int lane = threadIdx.x & 31;
    const int row  = blockIdx.x * 8 + wave;   // grid sized exactly, no tail
    const float* p = src + (size_t)row * HEAD_D;
    float x0 = p[lane * 2];
    float x1 = p[lane * 2 + 1];
    float s  = x0 * x0 + x1 * x1;
    #pragma unroll
    for (int m = 16; m >= 1; m >>= 1) s += __shfl_xor(s, m, 32);
    const float inv = __builtin_amdgcn_rsqf(s);   // v_rsq_f32, ~1 ulp
    _Float16* q = dst + (size_t)row * HEAD_D;
    q[lane * 2]     = (_Float16)(x0 * inv);
    q[lane * 2 + 1] = (_Float16)(x1 * inv);
    if (lane == 0) sq_out[row] = s;
}

// ---------------------------------------------------------------------------
// Kernel 0b: V f32 -> f16 copy-convert (4 elements / thread, vectorized).
// ---------------------------------------------------------------------------
__global__ __launch_bounds__(256)
void pka_v_to_f16(const float* __restrict__ src, _Float16* __restrict__ dst) {
    const size_t i = ((size_t)blockIdx.x * 256 + threadIdx.x) * 4;
    const float4 x = *(const float4*)(src + i);
    _Float16 h[4] = {(_Float16)x.x, (_Float16)x.y, (_Float16)x.z, (_Float16)x.w};
    *(uint64_t*)(dst + i) = *(const uint64_t*)h;   // single b64 store
}

// ---------------------------------------------------------------------------
// Main fused kernel: one workgroup = one (bh, 32-row q-tile).
//   Phase 1: S = qn . kn^T via v_wmma_f32_16x16x32_f16 -> LDS (32 x 2048 f32).
//   Phase 2: softmax from a register-cached row; attn -> HBM once; f16 in LDS.
//   Phase 3: out = attn @ V via v_wmma_f32_16x16x32_f16, pure f16 operands.
// ---------------------------------------------------------------------------
__global__ __launch_bounds__(256)
void pka_main(const _Float16* __restrict__ qn, const _Float16* __restrict__ kn,
              const float* __restrict__ qsq, const float* __restrict__ ksq,
              const _Float16* __restrict__ vh,
              float* __restrict__ out, float* __restrict__ attn) {
    extern __shared__ float sc[];           // [QTILE][LDS_STRIDE]

    const int tid  = threadIdx.x;
    const int lane = tid & 31;
    const int wave = tid >> 5;              // 0..7
    const int l15  = lane & 15;
    const int half = lane >> 4;             // 0 or 1
    const int qt   = blockIdx.x;            // 0..63  (q tile)
    const int bh   = blockIdx.y;            // 0..31
    const int q0   = qt * QTILE;

    const size_t rowbase = (size_t)bh * S_LEN;
    const _Float16* qn_bh = qn + rowbase * HEAD_D;
    const _Float16* kn_bh = kn + rowbase * HEAD_D;
    const _Float16* v_bh  = vh + rowbase * HEAD_D;
    const float*    qsq_b = qsq + rowbase;
    const float*    ksq_b = ksq + rowbase;

    const float inv_sqrt_d = 0.125f;        // 1/sqrt(64)

    // ---------------- Phase 1: scores ----------------
    {
        const int mtile = wave & 1;         // which 16 q-rows
        const int ngrp  = wave >> 1;        // 4 N-groups, interleaved
        const int arow  = q0 + mtile * 16 + l15;

        // A fragments (reused across all 128 N tiles): K 0..31 and 32..63
        const _Float16* ap = qn_bh + (size_t)arow * HEAD_D;
        const v16h a0 = load_fragA(ap, half);
        const v16h a1 = load_fragA(ap + 32, half);

        // per-row magnitude term, hoisted: qsq/2sqrt(d) for this lane's 8 M rows
        float qmag[8];
        #pragma unroll
        for (int j = 0; j < 8; ++j)
            qmag[j] = qsq_b[q0 + mtile * 16 + half * 8 + j] * (0.5f * inv_sqrt_d);

        for (int i = 0; i < 32; ++i) {      // 128 N tiles / 4 groups
            const int nt  = ngrp + 4 * i;
            const int col = nt * 16 + l15;  // key index
            const _Float16* bp = kn_bh + (size_t)col * HEAD_D;
            const v16h b0 = load_fragB(bp, half);
            const v16h b1 = load_fragB(bp + 32, half);
            v8f c = {};
            c = __builtin_amdgcn_wmma_f32_16x16x32_f16(false, a0, false, b0,
                                                       (short)0, c, false, false);
            c = __builtin_amdgcn_wmma_f32_16x16x32_f16(false, a1, false, b1,
                                                       (short)0, c, false, false);
            const float kmag = ksq_b[col] * (0.5f * inv_sqrt_d);
            #pragma unroll
            for (int j = 0; j < 8; ++j) {
                const int mloc = mtile * 16 + half * 8 + j;   // C layout
                const float arg = fmaxf(TWO_APPROX - 2.0f * c[j], 0.0f);
                // sin(pi*sqrt(arg)): v_sin_f32 takes revolutions -> 0.5*sqrt
                const float sn  = __builtin_amdgcn_sinf(
                                      0.5f * __builtin_amdgcn_sqrtf(arg));
                sc[mloc * LDS_STRIDE + nt * 16 + l15] =
                    fmaf(sn * sn, -2.0f * inv_sqrt_d, qmag[j] + kmag);
            }
        }
    }
    __syncthreads();

    // ---- Phase 2: softmax (one LDS read per element), attn write, f16 pack
    {
        float* attn_bh = attn + rowbase * S_LEN;
        for (int rr = 0; rr < 4; ++rr) {
            const int r = wave * 4 + rr;
            float* row = sc + r * LDS_STRIDE;
            _Float16* row16 = (_Float16*)row;     // in-place f16 pack
            float x[64];                          // register-cached row slice
            #pragma unroll
            for (int i = 0; i < 64; ++i) x[i] = row[lane + 32 * i];
            float m = x[0];
            #pragma unroll
            for (int i = 1; i < 64; ++i) m = fmaxf(m, x[i]);
            #pragma unroll
            for (int o = 16; o >= 1; o >>= 1) m = fmaxf(m, __shfl_xor(m, o, 32));
            float s = 0.0f;
            #pragma unroll
            for (int i = 0; i < 64; ++i) { x[i] = __expf(x[i] - m); s += x[i]; }
            #pragma unroll
            for (int o = 16; o >= 1; o >>= 1) s += __shfl_xor(s, o, 32);
            const float inv = __builtin_amdgcn_rcpf(s);   // v_rcp_f32
            float* arow = attn_bh + (size_t)(q0 + r) * S_LEN;
            #pragma unroll
            for (int i = 0; i < 64; ++i) {
                const int k = lane + 32 * i;
                const float e = x[i] * inv;
                arow[k]  = e;                 // single coalesced HBM write
                // safe in-place alias: half k lands in word k/2 (<=k), and the
                // f32 row was fully consumed into registers above.
                row16[k] = (_Float16)e;
            }
        }
    }
    __syncthreads();

    // ---------------- Phase 3: out = attn @ V (all-f16 operands) ----------
    {
        const int mtile = wave >> 2;        // 0/1 : 16 q-rows
        const int ntile = wave & 3;         // 0..3: 16 head-dim cols
        const int col   = ntile * 16 + l15;
        const _Float16* arow16 =
            (const _Float16*)(sc + (mtile * 16 + l15) * LDS_STRIDE);
        v8f c = {};
        for (int kk = 0; kk < 64; ++kk) {   // K = 2048 in chunks of 32
            const int kb = kk * 32;
            const v16h a = load_fragA(arow16 + kb, half);  // 2x ds_load_b128
            v16h b;
            #pragma unroll
            for (int v = 0; v < 8; ++v) {   // strided V columns: f16 loads
                const int kv = kb + 8 * half + 2 * v + (v >= 4 ? 8 : 0);
                b[2 * v]     = v_bh[(size_t)kv * HEAD_D + col];
                b[2 * v + 1] = v_bh[(size_t)(kv + 1) * HEAD_D + col];
            }
            c = __builtin_amdgcn_wmma_f32_16x16x32_f16(false, a, false, b,
                                                       (short)0, c, false, false);
        }
        float* orow = out + (rowbase + q0) * HEAD_D;
        #pragma unroll
        for (int j = 0; j < 8; ++j) {
            const int mloc = mtile * 16 + half * 8 + j;
            orow[(size_t)mloc * HEAD_D + ntile * 16 + l15] = c[j];
        }
    }
}

// ---------------------------------------------------------------------------
// Host launch. Workspace layout (25.7 MB):
//   [0)            qn  f16  65536*64        = 8,388,608 B
//   [8388608)      kn  f16  65536*64        = 8,388,608 B
//   [16777216)     vh  f16  65536*64        = 8,388,608 B
//   [25165824)     qsq f32  65536           =   262,144 B
//   [25427968)     ksq f32  65536           =   262,144 B
// ---------------------------------------------------------------------------
extern "C" void kernel_launch(void* const* d_in, const int* in_sizes, int n_in,
                              void* d_out, int out_size, void* d_ws, size_t ws_size,
                              hipStream_t stream) {
    const float* q = (const float*)d_in[0];
    const float* k = (const float*)d_in[1];
    const float* v = (const float*)d_in[2];

    float* out  = (float*)d_out;                         // [4,8,2048,64]
    float* attn = out + (size_t)BH_CNT * S_LEN * HEAD_D; // [4,8,2048,2048]

    char* ws = (char*)d_ws;
    _Float16* qn  = (_Float16*)(ws);
    _Float16* kn  = (_Float16*)(ws + 8388608);
    _Float16* vh  = (_Float16*)(ws + 16777216);
    float*    qsq = (float*)(ws + 25165824);
    float*    ksq = (float*)(ws + 25427968);

    pka_norm_precompute<<<NROWS / 8, 256, 0, stream>>>(q, qn, qsq);
    pka_norm_precompute<<<NROWS / 8, 256, 0, stream>>>(k, kn, ksq);
    pka_v_to_f16<<<(NROWS * HEAD_D) / (256 * 4), 256, 0, stream>>>(v, vh);

    dim3 grid(S_LEN / QTILE, BH_CNT);                    // (64, 32)
    size_t lds_bytes = (size_t)QTILE * LDS_STRIDE * sizeof(float); // 262,656 B
    pka_main<<<grid, 256, lds_bytes, stream>>>(qn, kn, qsq, ksq, vh, out, attn);
}
